// MultiHeadAttention_17248588660842
// MI455X (gfx1250) — matrix operand order repaired and checked by reading, not compiled, not run on hardware
//
#include <hip/hip_runtime.h>

// ---------------------------------------------------------------------------
// MI455X (gfx1250) fused MHA: bf16 WMMA GEMMs + in-register RoPE + flash attn
// Round 4: fix async-LDS builtin pointer types (int4 addrspace(1/3) pointers).
// ---------------------------------------------------------------------------

#define BATCH    2
#define SEQ      4096
#define DMODEL   512
#define NHEADS   8
#define HDIM     64
#define MROWS    (BATCH * SEQ)          // 8192

typedef __attribute__((ext_vector_type(16))) __bf16 v16bf;
typedef __attribute__((ext_vector_type(8)))  float  v8f;
typedef __attribute__((ext_vector_type(4)))  __bf16 bf4;
typedef __attribute__((ext_vector_type(4)))  int    v4i;

union FragU { uint4 u[2]; v16bf v; };

// ---------------------------------------------------------------------------
// Async global->LDS copy (16 B per lane). gfx1250 ASYNCcnt path when available.
// Builtin signature (from clang diagnostic): (v4i addrspace(1)*, v4i addrspace(3)*, Ii, Ii)
// ---------------------------------------------------------------------------
#if defined(__has_builtin)
#if __has_builtin(__builtin_amdgcn_global_load_async_to_lds_b128)
#define ASYNC_LDS 1
#endif
#endif

#ifdef ASYNC_LDS
typedef __attribute__((address_space(1))) v4i gv4i;
typedef __attribute__((address_space(3))) v4i lv4i;
#endif

__device__ __forceinline__ void async_cp16(__bf16* l, const __bf16* g)
{
#ifdef ASYNC_LDS
    __builtin_amdgcn_global_load_async_to_lds_b128((gv4i*)g, (lv4i*)l, 0, 0);
#else
    *reinterpret_cast<uint4*>(l) = *reinterpret_cast<const uint4*>(g);
#endif
}

__device__ __forceinline__ void wait_async_prev()   // prev tile done, 8 newest may fly
{
#ifdef ASYNC_LDS
#if __has_builtin(__builtin_amdgcn_s_wait_asynccnt)
    __builtin_amdgcn_s_wait_asynccnt(8);
#else
    asm volatile("s_wait_asynccnt 0x8" ::: "memory");
#endif
#endif
}

__device__ __forceinline__ void wait_async_all()
{
#ifdef ASYNC_LDS
#if __has_builtin(__builtin_amdgcn_s_wait_asynccnt)
    __builtin_amdgcn_s_wait_asynccnt(0);
#else
    asm volatile("s_wait_asynccnt 0x0" ::: "memory");
#endif
#endif
}

// ---------------------------------------------------------------------------
// Stage 0: fp32 -> bf16 conversion (vectorized x4)
// ---------------------------------------------------------------------------
__global__ __launch_bounds__(256)
void cvt_f32_bf16_kernel(const float* __restrict__ src, __bf16* __restrict__ dst, int n4)
{
    int i = blockIdx.x * blockDim.x + threadIdx.x;
    if (i < n4) {
        float4 f = reinterpret_cast<const float4*>(src)[i];
        bf4 o = { (__bf16)f.x, (__bf16)f.y, (__bf16)f.z, (__bf16)f.w };
        reinterpret_cast<bf4*>(dst)[i] = o;
    }
}

// ---------------------------------------------------------------------------
// A-fragment (16x32 bf16, row-major source): lane m = lane&15, khalf = lane>>4
//   vgpr0-3: K = kbase + khalf*8 + {0..7}, vgpr4-7: K = kbase + 16 + khalf*8 + {0..7}
// ---------------------------------------------------------------------------
__device__ __forceinline__ v16bf load_frag_a(const __bf16* rowptr, int kbase, int khalf)
{
    FragU un;
    const __bf16* p = rowptr + kbase + khalf * 8;
    un.u[0] = *reinterpret_cast<const uint4*>(p);
    un.u[1] = *reinterpret_cast<const uint4*>(p + 16);
    return un.v;
}

// B-fragment (32x16 bf16): lane n = lane&15, khalf = lane>>4
//   vgpr0-3: K = kbase + khalf*8 + {0..7}, vgpr4-7: K = kbase + 16 + khalf*8 + {0..7}
__device__ __forceinline__ v16bf load_frag_b(const __bf16* colptr, int kbase, int khalf)
{
    FragU un;
    const __bf16* p = colptr + kbase + khalf * 8;
    un.u[0] = *reinterpret_cast<const uint4*>(p);
    un.u[1] = *reinterpret_cast<const uint4*>(p + 16);
    return un.v;
}

__device__ __forceinline__ v8f wmma_bf16(v16bf a, v16bf b, v8f c)
{
    return __builtin_amdgcn_wmma_f32_16x16x32_bf16(false, a, false, b, (short)0, c,
                                                   false, false);
}

// ---------------------------------------------------------------------------
// Stage 1: QKV projection (y = x @ W^T) + RoPE + layout shuffle.
//   grid = (MROWS/64, NHEADS, 3), block = 128 (4 waves, 16 rows each)
//   z=0 -> Q  [BH, S, 64]  (RoPE, pre-scaled by 1/sqrt(Hd))
//   z=1 -> K  [BH, S, 64]  (RoPE)
//   z=2 -> Vt [BH, 64, S]  (transposed for PV B-fragments)
// ---------------------------------------------------------------------------
__global__ __launch_bounds__(128)
void qkv_rope_kernel(const __bf16* __restrict__ xb,
                     const __bf16* __restrict__ Wq,
                     const __bf16* __restrict__ Wk,
                     const __bf16* __restrict__ Wv,
                     __bf16* __restrict__ qout,
                     __bf16* __restrict__ kout,
                     __bf16* __restrict__ vtout)
{
    const int lane  = threadIdx.x & 31;
    const int wave  = threadIdx.x >> 5;
    const int m     = lane & 15;
    const int khalf = lane >> 4;
    const int z     = blockIdx.z;
    const __bf16* W = (z == 0) ? Wq : (z == 1) ? Wk : Wv;

    const int row0 = blockIdx.x * 64 + wave * 16;   // global row = b*S + s
    const int h    = blockIdx.y;                    // head == 64-col group

    const __bf16* arow = xb + (size_t)(row0 + m) * DMODEL;
    const __bf16* brow[4];
#pragma unroll
    for (int nt = 0; nt < 4; ++nt)
        brow[nt] = W + (size_t)(h * 64 + nt * 16 + m) * DMODEL;

    v8f acc[4] = {};

    v16bf af = load_frag_a(arow, 0, khalf);
    v16bf bf[4];
#pragma unroll
    for (int nt = 0; nt < 4; ++nt) bf[nt] = load_frag_b(brow[nt], 0, khalf);

#pragma unroll
    for (int k0 = 0; k0 < DMODEL; k0 += 32) {
        v16bf afn;
        v16bf bfn[4];
        if (k0 + 32 < DMODEL) {
            afn = load_frag_a(arow, k0 + 32, khalf);
#pragma unroll
            for (int nt = 0; nt < 4; ++nt)
                bfn[nt] = load_frag_b(brow[nt], k0 + 32, khalf);
        }
#pragma unroll
        for (int nt = 0; nt < 4; ++nt)
            acc[nt] = wmma_bf16(af, bf[nt], acc[nt]);
        if (k0 + 32 < DMODEL) {
            af = afn;
#pragma unroll
            for (int nt = 0; nt < 4; ++nt) bf[nt] = bfn[nt];
        }
    }

    // Epilogue: C layout row = r + 8*khalf, col = nt*16 + (lane&15)
#pragma unroll
    for (int nt = 0; nt < 4; ++nt) {
        const int d = nt * 16 + m;            // head-dim position 0..63
        const int i2 = (d >> 1) * 2;          // 2*i
        const float freq = __expf(-(float)i2 * (9.210340371976184f / 64.0f)); // 10000^(-2i/64)
        const float sgn  = (d & 1) ? 1.0f : -1.0f;
#pragma unroll
        for (int r = 0; r < 8; ++r) {
            const int grow = row0 + r + 8 * khalf;
            const int b = grow >> 12;         // / SEQ
            const int s = grow & (SEQ - 1);
            float val = acc[nt][r];
            if (z < 2) {
                float sn, cs;
                __sincosf((float)s * freq, &sn, &cs);
                float other = __shfl_xor(val, 1, 32);        // partner of the pair
                float out = val * cs + sgn * other * sn;     // even: xe*c - xo*s ; odd: xe*s + xo*c
                if (z == 0) out *= 0.125f;                   // 1/sqrt(64) folded into Q
                __bf16* dst = (z == 0) ? qout : kout;
                dst[(((size_t)(b * NHEADS + h)) * SEQ + s) * HDIM + d] = (__bf16)out;
            } else {
                vtout[(((size_t)(b * NHEADS + h)) * HDIM + d) * SEQ + s] = (__bf16)val;
            }
        }
    }
}

// ---------------------------------------------------------------------------
// Stage 2: causal flash attention with async double-buffered K/V staging.
//   grid = (B*H, S/64), block = 128 (4 waves x 16 query rows)
//   q,k: [BH,S,64] bf16 (q pre-scaled), vt: [BH,64,S] bf16
//   aout: [B,S,512] bf16
// Per KV tile: async-copy next tile to LDS while current tile computes.
// ---------------------------------------------------------------------------
__global__ __launch_bounds__(128)
void flash_attn_kernel(const __bf16* __restrict__ q,
                       const __bf16* __restrict__ kk,
                       const __bf16* __restrict__ vt,
                       __bf16* __restrict__ aout)
{
    __shared__ __align__(16) __bf16 kbuf[2][64][HDIM];   // K tile  [kv][hd]
    __shared__ __align__(16) __bf16 vbuf[2][HDIM][64];   // Vt tile [hd][kv]
    __shared__ __align__(16) __bf16 pbuf[4][16 * 64];

    const int lane  = threadIdx.x & 31;
    const int wave  = threadIdx.x >> 5;
    const int m     = lane & 15;
    const int khalf = lane >> 4;
    const int bh    = blockIdx.x;
    const int b     = bh >> 3;
    const int h     = bh & 7;
    const int q0    = blockIdx.y * 64 + wave * 16;   // first query row of this wave

    const __bf16* kbase = kk + (size_t)bh * SEQ * HDIM;
    const __bf16* vbase = vt + (size_t)bh * HDIM * SEQ;

    // Stage one 64x64 K tile + 64x64 Vt tile: 512 chunks of 16B each, 8/thread.
    auto stage_tile = [&](int buf, int kv0) {
#pragma unroll
        for (int i = 0; i < 4; ++i) {
            const int c    = (int)threadIdx.x + i * 128;  // 0..511
            const int row  = c >> 3;                      // 0..63
            const int col8 = (c & 7) * 8;                 // bf16 elements
            async_cp16(&kbuf[buf][row][col8],
                       kbase + (size_t)(kv0 + row) * HDIM + col8);
            async_cp16(&vbuf[buf][row][col8],
                       vbase + (size_t)row * SEQ + kv0 + col8);
        }
    };

    // Resident Q fragments (two 32-wide K-steps over head dim)
    v16bf qf[2];
    const __bf16* qrow = q + ((size_t)bh * SEQ + q0 + m) * HDIM;
#pragma unroll
    for (int ks = 0; ks < 2; ++ks) qf[ks] = load_frag_a(qrow, ks * 32, khalf);

    v8f oacc[4] = {};
    float mrow[8], lrow[8];
#pragma unroll
    for (int r = 0; r < 8; ++r) { mrow[r] = -__builtin_inff(); lrow[r] = 0.0f; }

    __bf16* pb = pbuf[wave];
    const int kv_end  = q0 + 16;                     // per-wave causal bound
    const int kv_last = (blockIdx.y + 1) * 64;       // block-uniform staging bound

    stage_tile(0, 0);

    for (int kv0 = 0; kv0 < kv_last; kv0 += 64) {
        const int cur = (kv0 >> 6) & 1;
        const int nxt = cur ^ 1;
        const bool more = (kv0 + 64 < kv_last);

        if (more) stage_tile(nxt, kv0 + 64);         // prefetch next tile
        if (more) wait_async_prev(); else wait_async_all();
        __syncthreads();                             // cur tile visible block-wide

        if (kv0 < kv_end) {                          // wave-uniform compute guard
            // ---- S = Q K^T (8 WMMA), K fragments from LDS ----
            v8f sacc[4] = {};
#pragma unroll
            for (int nt = 0; nt < 4; ++nt) {
                const __bf16* krow = &kbuf[cur][nt * 16 + m][0];
#pragma unroll
                for (int ks = 0; ks < 2; ++ks)
                    sacc[nt] = wmma_bf16(qf[ks], load_frag_b(krow, ks * 32, khalf),
                                         sacc[nt]);
            }

            // ---- V fragments from LDS; softmax VALU below hides DS latency ----
            v16bf vf[2][4];
#pragma unroll
            for (int ks = 0; ks < 2; ++ks)
#pragma unroll
                for (int nt = 0; nt < 4; ++nt)
                    vf[ks][nt] = load_frag_b(&vbuf[cur][nt * 16 + m][0], ks * 32, khalf);

            // ---- causal mask + online softmax (row = r + 8*khalf, col = m) ----
#pragma unroll
            for (int r = 0; r < 8; ++r) {
                const int qi = q0 + r + 8 * khalf;
                float mx = -__builtin_inff();
#pragma unroll
                for (int nt = 0; nt < 4; ++nt) {
                    const int kv = kv0 + nt * 16 + m;
                    float sv = (kv <= qi) ? sacc[nt][r] : -__builtin_inff();
                    sacc[nt][r] = sv;
                    mx = fmaxf(mx, sv);
                }
#pragma unroll
                for (int off = 1; off < 16; off <<= 1)
                    mx = fmaxf(mx, __shfl_xor(mx, off, 32));

                const float mnew  = fmaxf(mrow[r], mx);
                const float alpha = __expf(mrow[r] - mnew);
                mrow[r] = mnew;

                float rs = 0.0f;
#pragma unroll
                for (int nt = 0; nt < 4; ++nt) {
                    float p = __expf(sacc[nt][r] - mnew);
                    sacc[nt][r] = p;
                    rs += p;
                }
#pragma unroll
                for (int off = 1; off < 16; off <<= 1)
                    rs += __shfl_xor(rs, off, 32);
                lrow[r] = lrow[r] * alpha + rs;
#pragma unroll
                for (int nt = 0; nt < 4; ++nt) oacc[nt][r] *= alpha;
            }

            // ---- P: C-layout -> A-fragment layout via LDS (bf16) ----
#pragma unroll
            for (int nt = 0; nt < 4; ++nt)
#pragma unroll
                for (int r = 0; r < 8; ++r)
                    pb[(r + 8 * khalf) * 64 + nt * 16 + m] = (__bf16)sacc[nt][r];

            asm volatile("s_wait_dscnt 0" ::: "memory");   // same-wave LDS RAW fence

            // ---- O += P V (8 WMMA) ----
#pragma unroll
            for (int ks = 0; ks < 2; ++ks) {
                v16bf pfrag = load_frag_a(pb + m * 64, ks * 32, khalf);
#pragma unroll
                for (int nt = 0; nt < 4; ++nt)
                    oacc[nt] = wmma_bf16(pfrag, vf[ks][nt], oacc[nt]);
            }
        }

        __syncthreads();     // all waves done reading cur before it is re-staged
    }

    // ---- normalize and emit [B, S, D] bf16 for the output projection ----
#pragma unroll
    for (int r = 0; r < 8; ++r) {
        const float inv = 1.0f / lrow[r];
        const int s = q0 + r + 8 * khalf;
#pragma unroll
        for (int nt = 0; nt < 4; ++nt) {
            float o = oacc[nt][r] * inv;
            aout[((size_t)b * SEQ + s) * DMODEL + h * 64 + nt * 16 + m] = (__bf16)o;
        }
    }
}

// ---------------------------------------------------------------------------
// Stage 3: output projection  out = attn @ Wo^T   (f32 output)
//   grid = (MROWS/64, DMODEL/64), block = 128
// ---------------------------------------------------------------------------
__global__ __launch_bounds__(128)
void out_proj_kernel(const __bf16* __restrict__ a,
                     const __bf16* __restrict__ Wo,
                     float* __restrict__ out)
{
    const int lane  = threadIdx.x & 31;
    const int wave  = threadIdx.x >> 5;
    const int m     = lane & 15;
    const int khalf = lane >> 4;
    const int row0  = blockIdx.x * 64 + wave * 16;
    const int ng    = blockIdx.y;

    const __bf16* arow = a + (size_t)(row0 + m) * DMODEL;
    const __bf16* brow[4];
#pragma unroll
    for (int nt = 0; nt < 4; ++nt)
        brow[nt] = Wo + (size_t)(ng * 64 + nt * 16 + m) * DMODEL;

    v8f acc[4] = {};

    v16bf af = load_frag_a(arow, 0, khalf);
    v16bf bf[4];
#pragma unroll
    for (int nt = 0; nt < 4; ++nt) bf[nt] = load_frag_b(brow[nt], 0, khalf);

#pragma unroll
    for (int k0 = 0; k0 < DMODEL; k0 += 32) {
        v16bf afn;
        v16bf bfn[4];
        if (k0 + 32 < DMODEL) {
            afn = load_frag_a(arow, k0 + 32, khalf);
#pragma unroll
            for (int nt = 0; nt < 4; ++nt)
                bfn[nt] = load_frag_b(brow[nt], k0 + 32, khalf);
        }
#pragma unroll
        for (int nt = 0; nt < 4; ++nt)
            acc[nt] = wmma_bf16(af, bf[nt], acc[nt]);
        if (k0 + 32 < DMODEL) {
            af = afn;
#pragma unroll
            for (int nt = 0; nt < 4; ++nt) bf[nt] = bfn[nt];
        }
    }

#pragma unroll
    for (int r = 0; r < 8; ++r) {
        const int grow = row0 + r + 8 * khalf;
#pragma unroll
        for (int nt = 0; nt < 4; ++nt)
            out[(size_t)grow * DMODEL + ng * 64 + nt * 16 + m] = acc[nt][r];
    }
}

// ---------------------------------------------------------------------------
// Host launcher
// ---------------------------------------------------------------------------
extern "C" void kernel_launch(void* const* d_in, const int* in_sizes, int n_in,
                              void* d_out, int out_size, void* d_ws, size_t ws_size,
                              hipStream_t stream)
{
    const float* x  = (const float*)d_in[0];
    const float* Wq = (const float*)d_in[1];
    const float* Wk = (const float*)d_in[2];
    const float* Wv = (const float*)d_in[3];
    const float* Wo = (const float*)d_in[4];
    float* out = (float*)d_out;

    char* ws = (char*)d_ws;
    const size_t XB   = (size_t)MROWS * DMODEL * 2;        // 8 MiB
    const size_t WB   = (size_t)DMODEL * DMODEL * 2;       // 512 KiB
    const size_t QKVB = (size_t)BATCH * NHEADS * SEQ * HDIM * 2; // 8 MiB each

    __bf16* xb  = (__bf16*)(ws);
    __bf16* wqb = (__bf16*)(ws + XB);
    __bf16* wkb = (__bf16*)(ws + XB + WB);
    __bf16* wvb = (__bf16*)(ws + XB + 2 * WB);
    __bf16* wob = (__bf16*)(ws + XB + 3 * WB);
    __bf16* qb  = (__bf16*)(ws + XB + 4 * WB);
    __bf16* kb  = (__bf16*)(ws + XB + 4 * WB + QKVB);
    __bf16* vtb = (__bf16*)(ws + XB + 4 * WB + 2 * QKVB);
    __bf16* ab  = (__bf16*)(ws + XB + 4 * WB + 3 * QKVB);

    // Stage 0: bf16 conversion
    {
        int n4x = (MROWS * DMODEL) / 4;       // 1,048,576
        int n4w = (DMODEL * DMODEL) / 4;      // 65,536
        cvt_f32_bf16_kernel<<<(n4x + 255) / 256, 256, 0, stream>>>(x,  xb,  n4x);
        cvt_f32_bf16_kernel<<<(n4w + 255) / 256, 256, 0, stream>>>(Wq, wqb, n4w);
        cvt_f32_bf16_kernel<<<(n4w + 255) / 256, 256, 0, stream>>>(Wk, wkb, n4w);
        cvt_f32_bf16_kernel<<<(n4w + 255) / 256, 256, 0, stream>>>(Wv, wvb, n4w);
        cvt_f32_bf16_kernel<<<(n4w + 255) / 256, 256, 0, stream>>>(Wo, wob, n4w);
    }

    // Stage 1: QKV + RoPE
    qkv_rope_kernel<<<dim3(MROWS / 64, NHEADS, 3), 128, 0, stream>>>(
        xb, wqb, wkb, wvb, qb, kb, vtb);

    // Stage 2: flash attention
    flash_attn_kernel<<<dim3(BATCH * NHEADS, SEQ / 64), 128, 0, stream>>>(
        qb, kb, vtb, ab);

    // Stage 3: output projection (f32 result)
    out_proj_kernel<<<dim3(MROWS / 64, DMODEL / 64), 128, 0, stream>>>(
        ab, wob, out);
}
